// InfNeRF_84482006713182
// MI455X (gfx1250) — compile-verified, matrix operand order checked by reading, not run on hardware
//
#include <hip/hip_runtime.h>
#include <hip/hip_bf16.h>

typedef _Float16 f16;
typedef __attribute__((ext_vector_type(16))) _Float16 v16h;
typedef __attribute__((ext_vector_type(8)))  _Float16 v8h;
typedef __attribute__((ext_vector_type(8)))  float    v8f;

#define NRAYS        16384
#define NSAMP        64
#define TNEAR        0.1f
#define TFAR         1.5f
#define TBLSIZE      524288

// Packed weight-tile offsets in d_ws (f16 units). Each B-tile is 512 f16
// ([32 lanes][16 f16], one v16h WMMA B operand per lane).
#define OFF_WG0  0        // 32x64 : 1x4 tiles
#define OFF_WG1  2048     // 64x64 : 2x4 tiles
#define OFF_WD   6144     // 64x16 : 2x1 tiles
#define OFF_WC0  7168     // 24x64 -> pad 32x64 : 1x4 tiles
#define OFF_WC1  9216     // 64x64 : 2x4
#define OFF_WC2  13312    // 64x64 : 2x4
#define OFF_WCH  17408    // 64x3  -> pad 64x16 : 2x1
#define W_TOTAL  18432

// ---------------- weight packing (runs once, tiny) ----------------
// B-operand layout for v_wmma_f32_16x16x32_f16 (16-bit, K=32, N=16):
//  lane L holds column n = L&15; kb = (L&16)?8:0;
//  element i -> k = kb + (i<8 ? i : i+8)   (i.e. K in {kb..kb+7, kb+16..kb+23})
__device__ __forceinline__ void pack_one(const float* W, int K, int N, int NT,
                                         f16* dst, int e) {
  int tile   = e >> 9;
  int within = e & 511;
  int lane   = within >> 4;
  int i      = within & 15;
  int kt = tile / NT, nt = tile % NT;
  int k = kt * 32 + ((lane & 16) ? 8 : 0) + (i < 8 ? i : i + 8);
  int n = nt * 16 + (lane & 15);
  float v = (k < K && n < N) ? W[k * N + n] : 0.0f;
  dst[e] = (f16)v;
}

__global__ void pack_weights_kernel(const float* Wg0, const float* Wg1,
                                    const float* Wd,  const float* Wc0,
                                    const float* Wc1, const float* Wc2,
                                    const float* Wch, f16* dst) {
  int e = blockIdx.x * blockDim.x + threadIdx.x;
  if      (e < OFF_WG1)  pack_one(Wg0, 32, 64, 4, dst + OFF_WG0, e - OFF_WG0);
  else if (e < OFF_WD)   pack_one(Wg1, 64, 64, 4, dst + OFF_WG1, e - OFF_WG1);
  else if (e < OFF_WC0)  pack_one(Wd,  64, 16, 1, dst + OFF_WD,  e - OFF_WD);
  else if (e < OFF_WC1)  pack_one(Wc0, 24, 64, 4, dst + OFF_WC0, e - OFF_WC0);
  else if (e < OFF_WC2)  pack_one(Wc1, 64, 64, 4, dst + OFF_WC1, e - OFF_WC1);
  else if (e < OFF_WCH)  pack_one(Wc2, 64, 64, 4, dst + OFF_WC2, e - OFF_WC2);
  else if (e < W_TOTAL)  pack_one(Wch, 64,  3, 1, dst + OFF_WCH, e - OFF_WCH);
}

// ---------------- WMMA helpers ----------------
#define WMMA_F16(a, b, c) \
  __builtin_amdgcn_wmma_f32_16x16x32_f16(false, (a), false, (b), (short)0, (c), false, false)

// Load A operand (16xK tile, row-major f16 in LDS) per ISA 16-bit A layout.
__device__ __forceinline__ v16h load_a_lds(const f16* base, int stride, int kofs, int lane) {
  int row = lane & 15;
  int kb  = kofs + ((lane & 16) ? 8 : 0);
  const f16* p0 = base + row * stride + kb;
  v8h lo = *(const v8h*)(p0);
  v8h hi = *(const v8h*)(p0 + 16);
  v16h a;
#pragma unroll
  for (int i = 0; i < 8; ++i) { a[i] = lo[i]; a[i + 8] = hi[i]; }
  return a;
}

// Load pre-packed B tile from global: one contiguous 32B read per lane.
__device__ __forceinline__ v16h load_b_glob(const f16* tile, int lane) {
  return *(const v16h*)(tile + lane * 16);
}

// C accumulator preloaded with bias (bias depends only on N = lane&15).
__device__ __forceinline__ v8f bias_c(const float* bias, int nofs, int lane) {
  float bv = bias[nofs + (lane & 15)];
  v8f c;
#pragma unroll
  for (int j = 0; j < 8; ++j) c[j] = bv;
  return c;
}

// Spill D tile (C/D layout: lane holds col n, VGPR j holds row mb+j) to
// row-major f16 LDS with ReLU.
__device__ __forceinline__ void store_d_relu(f16* base, int stride, int nofs,
                                             int lane, v8f c) {
  int n0 = lane & 15;
  int mb = (lane & 16) ? 8 : 0;
#pragma unroll
  for (int j = 0; j < 8; ++j) {
    float v = c[j] > 0.0f ? c[j] : 0.0f;
    base[(mb + j) * stride + nofs + n0] = (f16)v;
  }
}

// ---------------- main fused NeRF kernel ----------------
__global__ void __launch_bounds__(128)
nerf_fused_kernel(const float* __restrict__ rays_o, const float* __restrict__ rays_d,
                  const float* __restrict__ tables,
                  const float* __restrict__ bg0, const float* __restrict__ bg1,
                  const float* __restrict__ bd,  const float* __restrict__ bc0,
                  const float* __restrict__ bc1, const float* __restrict__ bc2,
                  const float* __restrict__ bch,
                  const f16*  __restrict__ wt,
                  float* __restrict__ out) {
  __shared__ __align__(16) f16 featA[64][32];        // hash features (all 64 samples)
  __shared__ __align__(16) f16 hbuf[4][2][16][64];   // per-wave activation ping-pong
  __shared__ __align__(16) f16 cbuf[4][16][32];      // per-wave color-MLP input
  __shared__ float dens_s[64];
  __shared__ float col_s[64][3];

  const int r    = blockIdx.x;
  const int tid  = threadIdx.x;
  const int wave = tid >> 5;
  const int lane = tid & 31;

  const float ox = rays_o[r * 3 + 0], oy = rays_o[r * 3 + 1], oz = rays_o[r * 3 + 2];
  const float dx = rays_d[r * 3 + 0], dy = rays_d[r * 3 + 1], dz = rays_d[r * 3 + 2];
  const float step = (TFAR - TNEAR) / (float)(NSAMP - 1);

  // ---- Phase A: hash-grid gather. 1024 (sample,level) tasks over 128 threads.
#pragma unroll
  for (int j = 0; j < 8; ++j) {
    int task = tid * 8 + j;
    int s = task >> 4;
    int l = task & 15;
    float z  = TNEAR + step * (float)s;
    float px = ox + dx * z, py = oy + dy * z, pz = oz + dz * z;
    float res = (float)(32 << l);
    float rm1 = res - 1.0f;
    float sx = fminf(fmaxf((px + 1.0f) * 0.5f * rm1, 0.0f), rm1);
    float sy = fminf(fmaxf((py + 1.0f) * 0.5f * rm1, 0.0f), rm1);
    float sz = fminf(fmaxf((pz + 1.0f) * 0.5f * rm1, 0.0f), rm1);
    float idx_f  = sx * res * res + sy * res + sz;
    float sizem1 = (l == 0) ? 32767.0f : (l == 1) ? 262143.0f : 524287.0f;
    int idx = (int)fminf(idx_f, sizem1);
    const float* tp = tables + ((size_t)l * TBLSIZE + (size_t)idx) * 2;
    featA[s][2 * l + 0] = (f16)tp[0];
    featA[s][2 * l + 1] = (f16)tp[1];
  }
  __syncthreads();

  // ---- Phase B: fused MLPs. Wave w owns samples [16w, 16w+16).
  f16* hA = &hbuf[wave][0][0][0];
  f16* hB = &hbuf[wave][1][0][0];
  f16* cb = &cbuf[wave][0][0];

  // L0: feat(16x32) @ Wg0(32x64) + bg0, ReLU -> hA
  {
    v16h a = load_a_lds(&featA[wave * 16][0], 32, 0, lane);
#pragma unroll
    for (int nt = 0; nt < 4; ++nt) {
      v8f c = bias_c(bg0, nt * 16, lane);
      c = WMMA_F16(a, load_b_glob(wt + OFF_WG0 + nt * 512, lane), c);
      store_d_relu(hA, 64, nt * 16, lane, c);
    }
  }
  // L1: hA(16x64) @ Wg1(64x64) + bg1, ReLU -> hB
  {
    v16h a0 = load_a_lds(hA, 64, 0, lane);
    v16h a1 = load_a_lds(hA, 64, 32, lane);
#pragma unroll
    for (int nt = 0; nt < 4; ++nt) {
      v8f c = bias_c(bg1, nt * 16, lane);
      c = WMMA_F16(a0, load_b_glob(wt + OFF_WG1 + (0 * 4 + nt) * 512, lane), c);
      c = WMMA_F16(a1, load_b_glob(wt + OFF_WG1 + (1 * 4 + nt) * 512, lane), c);
      store_d_relu(hB, 64, nt * 16, lane, c);
    }
  }
  // Ld: hB(16x64) @ Wd(64x16) + bd -> geo tile (registers)
  v8f geo;
  {
    v16h a0 = load_a_lds(hB, 64, 0, lane);
    v16h a1 = load_a_lds(hB, 64, 32, lane);
    geo = bias_c(bd, 0, lane);
    geo = WMMA_F16(a0, load_b_glob(wt + OFF_WD + 0,   lane), geo);
    geo = WMMA_F16(a1, load_b_glob(wt + OFF_WD + 512, lane), geo);
  }
  // density (col 0, softplus(x-1)) + geo_feat (cols 1..15 -> cbuf cols 0..14)
  {
    int n0 = lane & 15;
    int mb = (lane & 16) ? 8 : 0;
    if (n0 == 0) {
#pragma unroll
      for (int j = 0; j < 8; ++j) {
        float x = geo[j] - 1.0f;
        dens_s[wave * 16 + mb + j] = (x > 20.0f) ? x : __logf(1.0f + __expf(x));
      }
    } else {
#pragma unroll
      for (int j = 0; j < 8; ++j)
        cb[(mb + j) * 32 + (n0 - 1)] = (f16)geo[j];
    }
  }
  // SH direction features -> cbuf cols 15..23, zero-pad 24..31 (per-wave, per-row)
  if (lane < 16) {
    int m = lane;
    float shv[9] = {0.5f, dx, dy, dz, dx * dy, dx * dz, dy * dz,
                    dx * dx - dy * dy, 3.0f * dz * dz - 1.0f};
#pragma unroll
    for (int q = 0; q < 9; ++q) cb[m * 32 + 15 + q] = (f16)shv[q];
#pragma unroll
    for (int q = 24; q < 32; ++q) cb[m * 32 + q] = (f16)0.0f;
  }
  // c0: cbuf(16x32) @ Wc0p(32x64) + bc0, ReLU -> hA
  {
    v16h a = load_a_lds(cb, 32, 0, lane);
#pragma unroll
    for (int nt = 0; nt < 4; ++nt) {
      v8f c = bias_c(bc0, nt * 16, lane);
      c = WMMA_F16(a, load_b_glob(wt + OFF_WC0 + nt * 512, lane), c);
      store_d_relu(hA, 64, nt * 16, lane, c);
    }
  }
  // c1: hA @ Wc1 + bc1, ReLU -> hB
  {
    v16h a0 = load_a_lds(hA, 64, 0, lane);
    v16h a1 = load_a_lds(hA, 64, 32, lane);
#pragma unroll
    for (int nt = 0; nt < 4; ++nt) {
      v8f c = bias_c(bc1, nt * 16, lane);
      c = WMMA_F16(a0, load_b_glob(wt + OFF_WC1 + (0 * 4 + nt) * 512, lane), c);
      c = WMMA_F16(a1, load_b_glob(wt + OFF_WC1 + (1 * 4 + nt) * 512, lane), c);
      store_d_relu(hB, 64, nt * 16, lane, c);
    }
  }
  // c2: hB @ Wc2 + bc2, ReLU -> hA
  {
    v16h a0 = load_a_lds(hB, 64, 0, lane);
    v16h a1 = load_a_lds(hB, 64, 32, lane);
#pragma unroll
    for (int nt = 0; nt < 4; ++nt) {
      v8f c = bias_c(bc2, nt * 16, lane);
      c = WMMA_F16(a0, load_b_glob(wt + OFF_WC2 + (0 * 4 + nt) * 512, lane), c);
      c = WMMA_F16(a1, load_b_glob(wt + OFF_WC2 + (1 * 4 + nt) * 512, lane), c);
      store_d_relu(hA, 64, nt * 16, lane, c);
    }
  }
  // ch: hA(16x64) @ Wchp(64x16) + bch -> sigmoid color (cols 0..2)
  {
    v16h a0 = load_a_lds(hA, 64, 0, lane);
    v16h a1 = load_a_lds(hA, 64, 32, lane);
    int n0 = lane & 15;
    float bv = (n0 < 3) ? bch[n0] : 0.0f;
    v8f c;
#pragma unroll
    for (int j = 0; j < 8; ++j) c[j] = bv;
    c = WMMA_F16(a0, load_b_glob(wt + OFF_WCH + 0,   lane), c);
    c = WMMA_F16(a1, load_b_glob(wt + OFF_WCH + 512, lane), c);
    int mb = (lane & 16) ? 8 : 0;
    if (n0 < 3) {
#pragma unroll
      for (int j = 0; j < 8; ++j)
        col_s[wave * 16 + mb + j][n0] = 1.0f / (1.0f + __expf(-c[j]));
    }
  }
  __syncthreads();

  // ---- Phase C: volumetric integration along the ray (sequential cumprod).
  if (tid == 0) {
    float dnorm = sqrtf(dx * dx + dy * dy + dz * dz);
    float* out_rgb = out;
    float* out_dep = out + 3 * NRAYS;
    float* out_acc = out + 4 * NRAYS;
    float* out_w   = out + 5 * NRAYS;
    float T = 1.0f, r0 = 0.0f, r1 = 0.0f, r2 = 0.0f, dep = 0.0f, acc = 0.0f;
    for (int s = 0; s < NSAMP; ++s) {
      float z     = TNEAR + step * (float)s;
      float dist  = ((s < NSAMP - 1) ? step : 1e10f) * dnorm;
      float alpha = 1.0f - __expf(-dens_s[s] * dist);
      float w     = alpha * T;
      T *= (1.0f - alpha + 1e-10f);
      r0  += w * col_s[s][0];
      r1  += w * col_s[s][1];
      r2  += w * col_s[s][2];
      dep += w * z;
      acc += w;
      out_w[(size_t)r * NSAMP + s] = w;
    }
    out_rgb[r * 3 + 0] = r0;
    out_rgb[r * 3 + 1] = r1;
    out_rgb[r * 3 + 2] = r2;
    out_dep[r] = dep;
    out_acc[r] = acc;
  }
}

extern "C" void kernel_launch(void* const* d_in, const int* in_sizes, int n_in,
                              void* d_out, int out_size, void* d_ws, size_t ws_size,
                              hipStream_t stream) {
  const float* rays_o = (const float*)d_in[0];
  const float* rays_d = (const float*)d_in[1];
  const float* tables = (const float*)d_in[2];
  const float* Wg0 = (const float*)d_in[3];  const float* bg0 = (const float*)d_in[4];
  const float* Wg1 = (const float*)d_in[5];  const float* bg1 = (const float*)d_in[6];
  const float* Wd  = (const float*)d_in[7];  const float* bd  = (const float*)d_in[8];
  const float* Wc0 = (const float*)d_in[9];  const float* bc0 = (const float*)d_in[10];
  const float* Wc1 = (const float*)d_in[11]; const float* bc1 = (const float*)d_in[12];
  const float* Wc2 = (const float*)d_in[13]; const float* bc2 = (const float*)d_in[14];
  const float* Wch = (const float*)d_in[15]; const float* bch = (const float*)d_in[16];
  f16* wt = (f16*)d_ws;

  pack_weights_kernel<<<(W_TOTAL + 255) / 256, 256, 0, stream>>>(
      Wg0, Wg1, Wd, Wc0, Wc1, Wc2, Wch, wt);

  nerf_fused_kernel<<<NRAYS, 128, 0, stream>>>(
      rays_o, rays_d, tables, bg0, bg1, bd, bc0, bc1, bc2, bch, wt,
      (float*)d_out);
}
